// GCN_13322988552211
// MI455X (gfx1250) — compile-verified, matrix operand order checked by reading, not run on hardware
//
#include <hip/hip_runtime.h>

#define N_N 50000
#define N_PAD 50048            // 391 blocks * 128 rows: lets GEMM store unguarded
#define N_E 800000
#define SLOPE 0.2f

typedef __attribute__((ext_vector_type(16))) __bf16 v16bf;
typedef __attribute__((ext_vector_type(8)))  float  v8f;

__device__ __forceinline__ void atomAddF(float* p, float v) {
  __hip_atomic_fetch_add(p, v, __ATOMIC_RELAXED, __HIP_MEMORY_SCOPE_AGENT);
}

__global__ void zero_f32(float* __restrict__ p, int n) {
  int i = blockIdx.x * blockDim.x + threadIdx.x;
  if (i < n) p[i] = 0.0f;
}

__global__ void deg_accum(const int* __restrict__ dst, float* __restrict__ deg, int e) {
  int i = blockIdx.x * blockDim.x + threadIdx.x;
  if (i < e) atomAddF(&deg[dst[i]], 1.0f);
}

__global__ void dinv_self(float* __restrict__ dinv, float* __restrict__ selfn, int n) {
  int i = blockIdx.x * blockDim.x + threadIdx.x;
  if (i < n) {
    float r = rsqrtf(dinv[i] + 1.0f);
    dinv[i]  = r;
    selfn[i] = r * r;
  }
}

__global__ void edge_norm(const int* __restrict__ src, const int* __restrict__ dst,
                          const float* __restrict__ dinv, float* __restrict__ en, int e) {
  int i = blockIdx.x * blockDim.x + threadIdx.x;
  if (i < e) en[i] = dinv[src[i]] * dinv[dst[i]];
}

// Pre-swizzle W (dout x din, row-major) into per-(nt,kt,lane) contiguous 16-elem
// bf16 B-fragments matching the 16-bit 32x16 B layout (ISA 05_wmma 7.12.2).
__global__ void pack_w(const float* __restrict__ W, __bf16* __restrict__ wp,
                       int din, int dout) {
  int tid = blockIdx.x * blockDim.x + threadIdx.x;
  int total = (dout >> 4) * (din >> 5) * 512;
  if (tid >= total) return;
  int e    = tid & 15;
  int lane = (tid >> 4) & 31;
  int tile = tid >> 9;
  int nKt  = din >> 5;
  int nt   = tile / nKt;
  int kt   = tile - nt * nKt;
  int n = (nt << 4) + (lane & 15);
  int k = (kt << 5) + (e & 7) + ((e >> 3) << 4) + ((lane >> 4) << 3);
  wp[tid] = (__bf16)W[n * din + k];
}

// h = act(in) @ W^T via v_wmma_f32_16x16x32_bf16. din fixed at 256.
// Block = 256 threads = 8 waves; each wave computes a 16-row strip.
// Packed weights (DOUT*256 bf16 = 128KB/64KB) are staged in LDS once per
// block (320KB WGP LDS), so B fragments come from ds_load instead of a
// cache round-trip per WMMA. h has N_PAD rows -> stores are unguarded.
template<int DOUT, bool ACT>
__global__ __launch_bounds__(256) void gemm_wmma(const float* __restrict__ in,
                                                 const __bf16* __restrict__ wpk,
                                                 float* __restrict__ h, int nrows) {
  constexpr int NT     = DOUT / 16;
  constexpr int CHUNKS = NT * 8 * 64;        // 16-byte chunks of packed W
  extern __shared__ char smem[];
  // cooperative copy of packed weights into LDS
  {
    const float4* gsrc = (const float4*)wpk;
    float4*       ldst = (float4*)smem;
    #pragma unroll 4
    for (int i = threadIdx.x; i < CHUNKS; i += 256) ldst[i] = gsrc[i];
  }

  const int lane = threadIdx.x & 31;
  const int wid  = __builtin_amdgcn_readfirstlane(threadIdx.x >> 5);
  const int m0   = blockIdx.x * 128 + wid * 16;
  const int khalf = lane >> 4;
  int row = m0 + (lane & 15);
  int rc  = row < nrows ? row : nrows - 1;   // clamp loads, keep EXEC full
  const float* rp = in + (size_t)rc * 256;

  // Build all 8 A fragments (K=256) once: per kt two contiguous 8-float runs.
  v16bf afr[8];
  #pragma unroll
  for (int kt = 0; kt < 8; ++kt) {
    const int base = kt * 32 + khalf * 8;
    const float4 a0 = *(const float4*)(rp + base);
    const float4 a1 = *(const float4*)(rp + base + 4);
    const float4 b0 = *(const float4*)(rp + base + 16);
    const float4 b1 = *(const float4*)(rp + base + 20);
    float f[16] = {a0.x, a0.y, a0.z, a0.w, a1.x, a1.y, a1.z, a1.w,
                   b0.x, b0.y, b0.z, b0.w, b1.x, b1.y, b1.z, b1.w};
    #pragma unroll
    for (int j = 0; j < 16; ++j) {
      float v = f[j];
      if (ACT) v = (v >= 0.0f) ? v : SLOPE * v;
      afr[kt][j] = (__bf16)v;
    }
  }

  __syncthreads();                            // weights resident in LDS

  const v16bf* wv  = (const v16bf*)smem;
  const int    col = lane & 15;
  const int    rb  = m0 + khalf * 8;          // C layout: VGPR r -> M = r + 8*khalf

  #pragma unroll 2
  for (int nt = 0; nt < NT; ++nt) {
    v16bf bf[8];
    #pragma unroll
    for (int kt = 0; kt < 8; ++kt) bf[kt] = wv[(nt * 8 + kt) * 32 + lane];
    v8f acc = {};
    #pragma unroll
    for (int kt = 0; kt < 8; ++kt) {
      acc = __builtin_amdgcn_wmma_f32_16x16x32_bf16(
          /*neg_a=*/false, afr[kt], /*neg_b=*/false, bf[kt],
          /*c_mod=*/(short)0, acc, /*reuse_a=*/false, /*reuse_b=*/false);
    }
    float* hp = h + (size_t)rb * DOUT + nt * 16 + col;
    #pragma unroll
    for (int r = 0; r < 8; ++r) hp[(size_t)r * DOUT] = acc[r];   // unguarded
  }
}

// out = selfnorm[i] * h + bias   (self-loop term + bias), float4 vectorized
template<int DOUT>
__global__ void self_bias(const float* __restrict__ h, const float* __restrict__ selfn,
                          const float* __restrict__ bias, float* __restrict__ out, int n) {
  const int CH = DOUT / 4;
  int t = blockIdx.x * blockDim.x + threadIdx.x;
  if (t >= n * CH) return;
  int i = t / CH;
  int c = t - i * CH;
  float s = selfn[i];
  float4 hv = ((const float4*)h)[t];
  float4 bv = ((const float4*)bias)[c];
  float4 o  = {s * hv.x + bv.x, s * hv.y + bv.y, s * hv.z + bv.z, s * hv.w + bv.w};
  ((float4*)out)[t] = o;
}

// out[dst] += enorm[e] * h[src]  — gather+atomic scatter; arrays resident in L2.
template<int DOUT>
__global__ void edge_scatter(const int* __restrict__ src, const int* __restrict__ dst,
                             const float* __restrict__ en, const float* __restrict__ h,
                             float* out, int e) {
  const int CH = DOUT / 4;
  int t = blockIdx.x * blockDim.x + threadIdx.x;
  if (t >= e * CH) return;
  int ei = t / CH;
  int c  = t - ei * CH;
  float w = en[ei];
  int s = src[ei], d = dst[ei];
  float4 hv = ((const float4*)(h + (size_t)s * DOUT))[c];
  float* op = out + (size_t)d * DOUT + (size_t)c * 4;
  atomAddF(op + 0, w * hv.x);
  atomAddF(op + 1, w * hv.y);
  atomAddF(op + 2, w * hv.z);
  atomAddF(op + 3, w * hv.w);
}

extern "C" void kernel_launch(void* const* d_in, const int* in_sizes, int n_in,
                              void* d_out, int out_size, void* d_ws, size_t ws_size,
                              hipStream_t stream) {
  (void)in_sizes; (void)n_in; (void)out_size; (void)ws_size;
  const float* x  = (const float*)d_in[0];
  const int* ei   = (const int*)d_in[1];
  const int* src  = ei;
  const int* dst  = ei + N_E;
  const float* W[6]; const float* Bv[6];
  for (int l = 0; l < 6; ++l) {
    W[l]  = (const float*)d_in[2 + 2 * l];
    Bv[l] = (const float*)d_in[3 + 2 * l];
  }

  char* ws = (char*)d_ws;
  size_t off = 0;
  auto take = [&](size_t bytes) -> void* {
    void* p = ws + off;
    off += (bytes + 255) & ~(size_t)255;
    return p;
  };
  float*  dinv  = (float*) take((size_t)N_N * 4);
  float*  selfn = (float*) take((size_t)N_N * 4);
  float*  en    = (float*) take((size_t)N_E * 4);
  __bf16* wpk   = (__bf16*)take((size_t)6 * 65536 * 2);
  float*  X0    = (float*) take((size_t)N_PAD * 256 * 4);  // h buffer (padded rows)
  float*  X1    = (float*) take((size_t)N_PAD * 256 * 4);  // layer output buffer

  // ---- normalization terms (layer-invariant) ----
  zero_f32 <<<(N_N + 255) / 256, 256, 0, stream>>>(dinv, N_N);
  deg_accum<<<(N_E + 255) / 256, 256, 0, stream>>>(dst, dinv, N_E);
  dinv_self<<<(N_N + 255) / 256, 256, 0, stream>>>(dinv, selfn, N_N);
  edge_norm<<<(N_E + 255) / 256, 256, 0, stream>>>(src, dst, dinv, en, N_E);

  // ---- pre-swizzle weights to bf16 B-fragment layout ----
  for (int l = 0; l < 6; ++l) {
    int dout  = (l == 5) ? 128 : 256;
    int total = (dout / 16) * 8 * 512;
    pack_w<<<total / 256, 256, 0, stream>>>(W[l], wpk + (size_t)l * 65536, 256, dout);
  }

  const int    gGemm  = N_PAD / 128;              // 391, exact cover of padded rows
  const size_t lds256 = (size_t)16 * 8 * 1024;    // 128KB packed W in LDS
  const size_t lds128 = (size_t)8  * 8 * 1024;    // 64KB
  const int    gSB256 = (N_N * 64 + 255) / 256;
  const int    gSC256 = (N_E * 64 + 255) / 256;

  // ---- layer 1 (no input activation) ----
  gemm_wmma<256, false><<<gGemm, 256, lds256, stream>>>(x, wpk, X0, N_N);
  self_bias<256><<<gSB256, 256, 0, stream>>>(X0, selfn, Bv[0], X1, N_N);
  edge_scatter<256><<<gSC256, 256, 0, stream>>>(src, dst, en, X0, X1, N_E);

  // ---- layers 2..5 ----
  for (int l = 1; l < 5; ++l) {
    gemm_wmma<256, true><<<gGemm, 256, lds256, stream>>>(X1, wpk + (size_t)l * 65536, X0, N_N);
    self_bias<256><<<gSB256, 256, 0, stream>>>(X0, selfn, Bv[l], X1, N_N);
    edge_scatter<256><<<gSC256, 256, 0, stream>>>(src, dst, en, X0, X1, N_E);
  }

  // ---- layer 6 (dout = 128, output -> d_out) ----
  float* out = (float*)d_out;
  gemm_wmma<128, true><<<gGemm, 256, lds128, stream>>>(X1, wpk + (size_t)5 * 65536, X0, N_N);
  self_bias<128><<<(N_N * 32 + 255) / 256, 256, 0, stream>>>(X0, selfn, Bv[5], out, N_N);
  edge_scatter<128><<<(N_E * 32 + 255) / 256, 256, 0, stream>>>(src, dst, en, X0, out, N_E);
}
